// SparseMoELayer_43327630082095
// MI455X (gfx1250) — compile-verified, matrix operand order checked by reading, not run on hardware
//
#include <hip/hip_runtime.h>
#include <hip/hip_bf16.h>

// ---------------------------------------------------------------------------
// Sparse MoE (top-2 of 8 experts), D=1024, F=4096, T=4096 tokens.
// zero -> gate(+scatter) -> prefix-sum -> WMMA GEMM1 (relu) -> WMMA GEMM2
// (+gate-weighted atomic combine). bf16 WMMA 16x16x32, f32 accumulate.
// GEMM2 A tile staged with CDNA5 GLOBAL_LOAD_ASYNC_TO_LDS_B128 (+ASYNCcnt).
// ---------------------------------------------------------------------------

#define D_MODEL 1024
#define D_FF    4096
#define NEXP    8
#define NTOK    4096   // BATCH*SEQ

#define A_STRIDE 40    // ushorts per LDS A row (80B, 16B-aligned rows)
#define B_STRIDE 40    // ushorts per LDS B row (80B, 16B-aligned rows)

#define GAS __attribute__((address_space(1)))   // force global (not flat) loads

typedef __attribute__((ext_vector_type(16))) __bf16 v16bf;
typedef __attribute__((ext_vector_type(2)))  __bf16 v2bf;
typedef __attribute__((ext_vector_type(8)))  float  v8f;
typedef __attribute__((ext_vector_type(4)))  float  v4f;

typedef const float GAS* gcfp;
typedef const v4f   GAS* gcv4p;

struct Frag {
    union { v16bf v; unsigned u[8]; };
};

// round-to-nearest-even float -> bf16 bits
__device__ __forceinline__ unsigned short f2bf(float f) {
    unsigned u = __float_as_uint(f);
    u += 0x7FFFu + ((u >> 16) & 1u);
    return (unsigned short)(u >> 16);
}

#if defined(__has_builtin)
#if __has_builtin(__builtin_amdgcn_cvt_pk_bf16_f32)
#define HAVE_CVT_PK_BF16 1
#endif
#if __has_builtin(__builtin_amdgcn_perm)
#define HAVE_PERM 1
#endif
#endif

// pack two floats -> packed bf16 pair (lo in [15:0], hi in [31:16]), RNE
__device__ __forceinline__ unsigned pack_bf2(float lo, float hi) {
#if defined(HAVE_CVT_PK_BF16)
    v2bf p = __builtin_amdgcn_cvt_pk_bf16_f32(lo, hi);
    union { v2bf v; unsigned u; } c;
    c.v = p;
    return c.u;
#elif defined(HAVE_PERM)
    unsigned ul = __float_as_uint(lo);
    unsigned uh = __float_as_uint(hi);
    ul += 0x7FFFu + ((ul >> 16) & 1u);
    uh += 0x7FFFu + ((uh >> 16) & 1u);
    // bytes: [0,1] = ul[2,3] (sel 2,3), [2,3] = uh[2,3] (sel 6,7)
    return __builtin_amdgcn_perm(uh, ul, 0x07060302u);
#else
    return (unsigned)f2bf(lo) | ((unsigned)f2bf(hi) << 16);
#endif
}

// Load one wave-fragment (16 rows x 32 K, bf16) from an LDS tile stored as
// [row][k] ushorts. Layout per CDNA5 ISA 16-bit A matrix:
// lanes 0-15 hold K {0..7,16..23}, lanes 16-31 hold K {8..15,24..31},
// each VGPR = one even-aligned (k, k+1) pair. With 16B-aligned rows the
// 4-dword groups lower to ds_load_b128.
__device__ __forceinline__ void load_frag(Frag& f, const unsigned short* base,
                                          int rowStride, int row, int half) {
#pragma unroll
    for (int r = 0; r < 8; ++r) {
        int kk = (r & 3) * 2 + half * 8 + (r >> 2) * 16;
        f.u[r] = *(const unsigned*)(base + row * rowStride + kk);
    }
}

// ---------------------------------------------------------------------------
__global__ __launch_bounds__(256) void moe_zero(float* __restrict__ out,
                                                int* __restrict__ counts) {
    size_t i = (size_t)blockIdx.x * blockDim.x + threadIdx.x;
    if (i < (size_t)NTOK * D_MODEL) out[i] = 0.0f;
    if (i < NEXP) counts[i] = 0;
}

// one wave per token: gate logits -> softmax -> top-2 -> scatter to lists
__global__ __launch_bounds__(256) void moe_gate(const float* __restrict__ x,
                                                const float* __restrict__ Wg,
                                                int* __restrict__ counts,
                                                int* __restrict__ tokens,
                                                float* __restrict__ gatew) {
    const int wave = threadIdx.x >> 5;
    const int lane = threadIdx.x & 31;
    const int t = blockIdx.x * 8 + wave;
    if (t >= NTOK) return;

    float acc[NEXP];
#pragma unroll
    for (int e = 0; e < NEXP; ++e) acc[e] = 0.0f;

    const float* xr = x + (size_t)t * D_MODEL;
    for (int d = lane; d < D_MODEL; d += 32) {
        float xv = xr[d];
#pragma unroll
        for (int e = 0; e < NEXP; ++e) acc[e] += xv * Wg[d * NEXP + e];
    }
#pragma unroll
    for (int e = 0; e < NEXP; ++e) {
#pragma unroll
        for (int off = 16; off > 0; off >>= 1) acc[e] += __shfl_xor(acc[e], off, 32);
    }

    if (lane == 0) {
        float mx = acc[0];
#pragma unroll
        for (int e = 1; e < NEXP; ++e) mx = fmaxf(mx, acc[e]);
        float p[NEXP];
        float s = 0.0f;
#pragma unroll
        for (int e = 0; e < NEXP; ++e) { p[e] = __expf(acc[e] - mx); s += p[e]; }
        float inv_s = 1.0f / s;
#pragma unroll
        for (int e = 0; e < NEXP; ++e) p[e] *= inv_s;

        int i1 = 0;
#pragma unroll
        for (int e = 1; e < NEXP; ++e) if (p[e] > p[i1]) i1 = e;
        int i2 = (i1 == 0) ? 1 : 0;
#pragma unroll
        for (int e = 0; e < NEXP; ++e) if (e != i1 && p[e] > p[i2]) i2 = e;

        float inv = 1.0f / (p[i1] + p[i2]);
        float w1 = p[i1] * inv, w2 = p[i2] * inv;

        int pos1 = atomicAdd(&counts[i1], 1);
        tokens[i1 * NTOK + pos1] = t;
        gatew[i1 * NTOK + pos1] = w1;
        int pos2 = atomicAdd(&counts[i2], 1);
        tokens[i2 * NTOK + pos2] = t;
        gatew[i2 * NTOK + pos2] = w2;
    }
}

__global__ void moe_offsets(const int* __restrict__ counts, int* __restrict__ offsets) {
    if (threadIdx.x == 0) {
        int run = 0;
        for (int e = 0; e < NEXP; ++e) { offsets[e] = run; run += counts[e]; }
    }
}

// ---------------------------------------------------------------------------
// GEMM1: h[slot, f] = relu(x[token, :] @ W1[e] + b1[e]) ; bf16 output
// block tile: 64 tokens x 128 F, K-step 32, 8 waves, wmma bf16 16x16x32
// ---------------------------------------------------------------------------
__global__ __launch_bounds__(256) void moe_ffn1(const float* __restrict__ x,
                                                const float* __restrict__ W1,
                                                const float* __restrict__ b1,
                                                const int* __restrict__ counts,
                                                const int* __restrict__ offsets,
                                                const int* __restrict__ tokens,
                                                unsigned short* __restrict__ hbuf) {
    __shared__ __attribute__((aligned(16))) unsigned short As[64][A_STRIDE];
    __shared__ __attribute__((aligned(16))) unsigned short Bs[128][B_STRIDE];
    __shared__ int tok[64];

    const int e    = blockIdx.z;
    const int cnt  = counts[e];
    const int row0 = blockIdx.y * 64;
    if (row0 >= cnt) return;                       // uniform exit
    const int f0   = blockIdx.x * 128;

    const int tid  = threadIdx.x;
    const int lane = tid & 31;
    const int wave = tid >> 5;
    const int mrow = (wave & 3) * 16;
    const int ncol = (wave >> 2) * 64;
    const int half = lane >> 4;
    const int lidx = lane & 15;

    if (tid < 64) {
        int r = row0 + tid;
        if (r >= cnt) r = cnt - 1;                 // clamp: rows masked in epilogue
        tok[tid] = tokens[e * NTOK + r];
    }
    __syncthreads();

    // hoisted global (addrspace-1) staging pointers
    gcfp      aptr[2];
    unsigned* adst[2];
#pragma unroll
    for (int i = 0; i < 2; ++i) {
        int idx = tid + i * 256;                   // 512 float4 total
        int m = idx >> 3, kq = idx & 7;            // 8 quads per 32-wide row
        aptr[i] = (gcfp)(x + (size_t)tok[m] * D_MODEL + kq * 4);
        adst[i] = (unsigned*)&As[m][kq * 4];
    }
    gcfp            bptr[4];
    unsigned short* bdst[4];
#pragma unroll
    for (int i = 0; i < 4; ++i) {
        int idx = tid + i * 256;                   // 1024 float4 total
        int k = idx >> 5, nq = idx & 31;           // 32 quads per k-row
        bptr[i] = (gcfp)(W1 + ((size_t)e * D_MODEL + k) * D_FF + f0 + nq * 4);
        bdst[i] = &Bs[nq * 4][k];
    }
    const float* bpf = &W1[((size_t)e * D_MODEL + 32) * D_FF + f0 + (tid & 127)];

    v8f acc[4] = {};

    for (int k0 = 0; k0 < D_MODEL; k0 += 32) {
        // stage A: 64x32 gathered token rows, fp32 -> bf16 (global b128 loads)
#pragma unroll
        for (int i = 0; i < 2; ++i) {
            v4f v = *(gcv4p)aptr[i];
            aptr[i] += 32;
            adst[i][0] = pack_bf2(v.x, v.y);
            adst[i][1] = pack_bf2(v.z, v.w);
        }
        // stage B: 32x128 of W1, fp32 -> bf16, stored [n][k] (global b128 loads)
#pragma unroll
        for (int i = 0; i < 4; ++i) {
            v4f v = *(gcv4p)bptr[i];
            bptr[i] += (size_t)32 * D_FF;
            unsigned short* p = bdst[i];
            p[0 * B_STRIDE] = f2bf(v.x);
            p[1 * B_STRIDE] = f2bf(v.y);
            p[2 * B_STRIDE] = f2bf(v.z);
            p[3 * B_STRIDE] = f2bf(v.w);
        }
        if (k0 + 32 < D_MODEL) __builtin_prefetch(bpf, 0, 1);
        bpf += (size_t)32 * D_FF;
        __syncthreads();

        Frag a;
        load_frag(a, &As[0][0], A_STRIDE, mrow + lidx, half);
#pragma unroll
        for (int t4 = 0; t4 < 4; ++t4) {
            Frag b;
            load_frag(b, &Bs[0][0], B_STRIDE, ncol + t4 * 16 + lidx, half);
            acc[t4] = __builtin_amdgcn_wmma_f32_16x16x32_bf16(
                false, a.v, false, b.v, (short)0, acc[t4], false, false);
        }
        __syncthreads();
    }

    // epilogue: bias + relu -> bf16 h rows at compact slots
    const int base = offsets[e] + row0;
#pragma unroll
    for (int t4 = 0; t4 < 4; ++t4) {
        int n = f0 + ncol + t4 * 16 + lidx;
        float bias = b1[e * D_FF + n];
#pragma unroll
        for (int v = 0; v < 8; ++v) {
            int m = mrow + v + half * 8;
            if (row0 + m < cnt) {
                float val = acc[t4][v] + bias;
                val = val > 0.0f ? val : 0.0f;
                hbuf[(size_t)(base + m) * D_FF + n] = f2bf(val);
            }
        }
    }
}

// ---------------------------------------------------------------------------
// GEMM2: y[slot, d] = h[slot, :] @ W2[e] + b2[e]; out[t,d] += w * y  (atomic)
// A tile is already bf16 & contiguous -> async global->LDS B128 copy.
// ---------------------------------------------------------------------------
__global__ __launch_bounds__(256) void moe_ffn2(const unsigned short* __restrict__ hbuf,
                                                const float* __restrict__ W2,
                                                const float* __restrict__ b2,
                                                const int* __restrict__ counts,
                                                const int* __restrict__ offsets,
                                                const int* __restrict__ tokens,
                                                const float* __restrict__ gatew,
                                                float* __restrict__ out) {
    __shared__ __attribute__((aligned(16))) unsigned short As[64][A_STRIDE];
    __shared__ __attribute__((aligned(16))) unsigned short Bs[128][B_STRIDE];
    __shared__ int   tok[64];
    __shared__ float twt[64];

    const int e    = blockIdx.z;
    const int cnt  = counts[e];
    const int row0 = blockIdx.y * 64;
    if (row0 >= cnt) return;                       // uniform exit
    const int n0   = blockIdx.x * 128;

    const int tid  = threadIdx.x;
    const int lane = tid & 31;
    const int wave = tid >> 5;
    const int mrow = (wave & 3) * 16;
    const int ncol = (wave >> 2) * 64;
    const int half = lane >> 4;
    const int lidx = lane & 15;

    if (tid < 64) {
        int r = row0 + tid;
        bool ok = (r < cnt);
        tok[tid] = ok ? tokens[e * NTOK + r] : -1;
        twt[tid] = ok ? gatew[e * NTOK + r] : 0.0f;
    }
    __syncthreads();

    // async A-copy assignment: 16 bytes (8 bf16) per thread per K-step
    const int cm = tid >> 2;                        // row 0..63
    const int cq = tid & 3;                         // 16B quad within 64B row
    size_t   crow = (size_t)(offsets[e] + row0 + cm);
    if (crow > (size_t)2 * NTOK - 1) crow = (size_t)2 * NTOK - 1;  // clamp tail
    const unsigned lds_dst =
        (unsigned)(size_t)&As[cm][cq * 8];          // 16B-aligned (row stride 80B)
    const unsigned short* gp = hbuf + crow * D_FF + cq * 8;

    gcfp            bptr[4];
    unsigned short* bdst[4];
#pragma unroll
    for (int i = 0; i < 4; ++i) {
        int idx = tid + i * 256;
        int k = idx >> 5, nq = idx & 31;
        bptr[i] = (gcfp)(W2 + ((size_t)e * D_FF + k) * D_MODEL + n0 + nq * 4);
        bdst[i] = &Bs[nq * 4][k];
    }
    const float* bpf = &W2[((size_t)e * D_FF + 32) * D_MODEL + n0 + (tid & 127)];

    v8f acc[4] = {};

    for (int k0 = 0; k0 < D_FF; k0 += 32) {
        // stage A: async copy 64x32 bf16 tile (one B128 per lane)
        asm volatile("global_load_async_to_lds_b128 %0, %1, off"
                     :: "v"(lds_dst), "v"(gp)
                     : "memory");
        gp += 32;
        // stage B: 32x128 of W2, fp32 -> bf16, stored [n][k] (global b128 loads)
#pragma unroll
        for (int i = 0; i < 4; ++i) {
            v4f v = *(gcv4p)bptr[i];
            bptr[i] += (size_t)32 * D_MODEL;
            unsigned short* p = bdst[i];
            p[0 * B_STRIDE] = f2bf(v.x);
            p[1 * B_STRIDE] = f2bf(v.y);
            p[2 * B_STRIDE] = f2bf(v.z);
            p[3 * B_STRIDE] = f2bf(v.w);
        }
        if (k0 + 32 < D_FF) __builtin_prefetch(bpf, 0, 1);
        bpf += (size_t)32 * D_MODEL;
        asm volatile("s_wait_asynccnt 0" ::: "memory");
        __syncthreads();

        Frag a;
        load_frag(a, &As[0][0], A_STRIDE, mrow + lidx, half);
#pragma unroll
        for (int t4 = 0; t4 < 4; ++t4) {
            Frag b;
            load_frag(b, &Bs[0][0], B_STRIDE, ncol + t4 * 16 + lidx, half);
            acc[t4] = __builtin_amdgcn_wmma_f32_16x16x32_bf16(
                false, a.v, false, b.v, (short)0, acc[t4], false, false);
        }
        __syncthreads();
    }

    // epilogue: bias, scale by gate weight, atomic combine (exactly 2 adds/elem)
#pragma unroll
    for (int t4 = 0; t4 < 4; ++t4) {
        int n = n0 + ncol + t4 * 16 + lidx;
        float bias = b2[e * D_MODEL + n];
#pragma unroll
        for (int v = 0; v < 8; ++v) {
            int m = mrow + v + half * 8;
            int tk = tok[m];
            if (tk >= 0) {
                float val = (acc[t4][v] + bias) * twt[m];
                atomicAdd(&out[(size_t)tk * D_MODEL + n], val);
            }
        }
    }
}

// ---------------------------------------------------------------------------
extern "C" void kernel_launch(void* const* d_in, const int* in_sizes, int n_in,
                              void* d_out, int out_size, void* d_ws, size_t ws_size,
                              hipStream_t stream) {
    const float* x  = (const float*)d_in[0];
    const float* Wg = (const float*)d_in[1];
    const float* W1 = (const float*)d_in[2];
    const float* b1 = (const float*)d_in[3];
    const float* W2 = (const float*)d_in[4];
    const float* b2 = (const float*)d_in[5];
    float* out = (float*)d_out;

    // workspace layout
    int*   counts  = (int*)d_ws;
    int*   offsets = counts + NEXP;
    int*   tokens  = offsets + NEXP;                         // NEXP*NTOK ints
    float* gatew   = (float*)(tokens + NEXP * NTOK);         // NEXP*NTOK floats
    unsigned short* hbuf = (unsigned short*)(gatew + NEXP * NTOK); // 2*NTOK x D_FF bf16

    // 1) zero output + counters
    moe_zero<<<((size_t)NTOK * D_MODEL + 255) / 256, 256, 0, stream>>>(out, counts);
    // 2) gating + routing scatter
    moe_gate<<<NTOK / 8, 256, 0, stream>>>(x, Wg, counts, tokens, gatew);
    // 3) prefix sum of expert counts
    moe_offsets<<<1, 32, 0, stream>>>(counts, offsets);
    // 4) expert FFN layer 1 (relu)
    dim3 g1(D_FF / 128, NTOK / 64, NEXP);
    moe_ffn1<<<g1, 256, 0, stream>>>(x, W1, b1, counts, offsets, tokens, hbuf);
    // 5) expert FFN layer 2 + weighted combine
    dim3 g2(D_MODEL / 128, NTOK / 64, NEXP);
    moe_ffn2<<<g2, 256, 0, stream>>>(hbuf, W2, b2, counts, offsets, tokens, gatew, out);
}
